// PairformerNoSeqLayer_20306605375687
// MI455X (gfx1250) — matrix operand order repaired and checked by reading, not compiled, not run on hardware
//
#include <hip/hip_runtime.h>
#include <hip/hip_bf16.h>

typedef unsigned short u16;
typedef __attribute__((ext_vector_type(16))) __bf16 v16bf;
typedef __attribute__((ext_vector_type(8)))  float  v8f;

#define DEV __device__ __forceinline__

DEV u16 f2bf(float f) {
    unsigned u = __float_as_uint(f);
    unsigned r = u + 0x7fffu + ((u >> 16) & 1u);   // round-to-nearest-even
    return (u16)(r >> 16);
}
DEV float bf2f(u16 h) { return __uint_as_float(((unsigned)h) << 16); }
DEV float sigm(float x) { return 1.0f / (1.0f + __expf(-x)); }

union FragU { uint4 u[2]; v16bf v; };

// ---- WMMA operand loaders (CDNA5 16-bit layouts, wave32) ----
// A 16x32: lane L holds row (base + L&15), K = (L>>4)*16 .. +15 (contiguous)
DEV v16bf load_a(const u16* A, int lda, int row_base, int k0, int lane) {
    const u16* p = A + (size_t)(row_base + (lane & 15)) * lda + k0 + ((lane >> 4) * 16);
    FragU f; f.u[0] = *(const uint4*)p; f.u[1] = *(const uint4*)(p + 8);
    return f.v;
}
// B 32x16 (=W^T): lane L holds column (base + L&15), K = (L>>4)*16 .. +15 (contiguous in W row)
DEV v16bf load_b(const u16* W, int ldw, int col_base, int k0, int lane) {
    const u16* p = W + (size_t)(col_base + (lane & 15)) * ldw + k0 + ((lane >> 4) * 16);
    FragU f; f.u[0] = *(const uint4*)p; f.u[1] = *(const uint4*)(p + 8);
    return f.v;
}
DEV v8f wmma_bf(v16bf a, v16bf b, v8f c) {
    return __builtin_amdgcn_wmma_f32_16x16x32_bf16(false, a, false, b, (short)0, c, false, false);
}

// Generic wave tile: C1 = A1*W1^T (and C2 = A2*W2^T when DUAL). All row-major, K contiguous.
template<int MT, int NT, bool DUAL>
DEV void gemm_run(const u16* A1, const u16* A2, int lda,
                  const u16* W1, const u16* W2, int ldw,
                  int m0, int n0, int K, int lane,
                  v8f (&c1)[MT][NT], v8f (&c2)[MT][NT]) {
    for (int k0 = 0; k0 < K; k0 += 32) {
        v16bf a1[MT], a2[MT];
#pragma unroll
        for (int mt = 0; mt < MT; ++mt) {
            a1[mt] = load_a(A1, lda, m0 + mt * 16, k0, lane);
            if (DUAL) a2[mt] = load_a(A2, lda, m0 + mt * 16, k0, lane);
        }
#pragma unroll
        for (int nt = 0; nt < NT; ++nt) {
            v16bf b1 = load_b(W1, ldw, n0 + nt * 16, k0, lane);
#pragma unroll
            for (int mt = 0; mt < MT; ++mt) c1[mt][nt] = wmma_bf(a1[mt], b1, c1[mt][nt]);
            if (DUAL) {
                v16bf b2 = load_b(W2, ldw, n0 + nt * 16, k0, lane);
#pragma unroll
                for (int mt = 0; mt < MT; ++mt) c2[mt][nt] = wmma_bf(a2[mt], b2, c2[mt][nt]);
            }
        }
    }
}

// ------------------------------------------------------------------ utility kernels
__global__ __launch_bounds__(256) void k_f2bf(const float* in, u16* out, int n) {
    int i = blockIdx.x * 256 + threadIdx.x;
    if (i < n) out[i] = f2bf(in[i]);
}

__global__ __launch_bounds__(128) void k_ln(const float* __restrict__ x,
                                            const float* __restrict__ g,
                                            const float* __restrict__ b,
                                            u16* __restrict__ out) {
    const int row = blockIdx.x, t = threadIdx.x;
    float v = x[(size_t)row * 128 + t];
    __shared__ float red[4];
    float s = v;
    s += __shfl_xor(s, 1, 32); s += __shfl_xor(s, 2, 32); s += __shfl_xor(s, 4, 32);
    s += __shfl_xor(s, 8, 32); s += __shfl_xor(s, 16, 32);
    if ((t & 31) == 0) red[t >> 5] = s;
    __syncthreads();
    float mean = (red[0] + red[1] + red[2] + red[3]) * 0.0078125f;
    __syncthreads();
    float dv = v - mean;
    float s2 = dv * dv;
    s2 += __shfl_xor(s2, 1, 32); s2 += __shfl_xor(s2, 2, 32); s2 += __shfl_xor(s2, 4, 32);
    s2 += __shfl_xor(s2, 8, 32); s2 += __shfl_xor(s2, 16, 32);
    if ((t & 31) == 0) red[t >> 5] = s2;
    __syncthreads();
    float var = (red[0] + red[1] + red[2] + red[3]) * 0.0078125f;
    float y = dv * __frsqrt_rn(var + 1e-5f) * g[t] + b[t];
    out[(size_t)row * 128 + t] = f2bf(y);
}

__global__ __launch_bounds__(128) void k_transpose(const float* __restrict__ z, float* __restrict__ zt) {
    const int m = blockIdx.x, t = threadIdx.x;   // m = i*256+j -> write row j*256+i
    zt[((size_t)(((m & 255) << 8) | (m >> 8))) * 128 + t] = z[(size_t)m * 128 + t];
}

// tb[h,i,j] = sum_d xn[i,j,d] * wb[h,d]
__global__ __launch_bounds__(256) void k_tb(const u16* __restrict__ xn, const u16* __restrict__ wb,
                                            float* __restrict__ tb) {
    const int m = blockIdx.x * 256 + threadIdx.x;     // 65536 rows
    float acc[4] = {0.f, 0.f, 0.f, 0.f};
    for (int d = 0; d < 128; ++d) {
        float x = bf2f(xn[(size_t)m * 128 + d]);
#pragma unroll
        for (int h = 0; h < 4; ++h) acc[h] += x * bf2f(wb[h * 128 + d]);
    }
#pragma unroll
    for (int h = 0; h < 4; ++h) tb[(size_t)h * 65536 + m] = acc[h];
}

// ------------------------------------------------------------------ GEMM kernels
// tri-mul input projection: x = (xn@p_in^T)*sigmoid(xn@g_in^T)*mask, scattered to a_t/b_t [d][.][.]
template<int SWZ>
__global__ __launch_bounds__(32) void k_gemm_trimul_in(const u16* __restrict__ xn,
                                                       const u16* __restrict__ pin,
                                                       const u16* __restrict__ gin,
                                                       const float* __restrict__ mask,
                                                       u16* __restrict__ aT, u16* __restrict__ bT) {
    const int lane = threadIdx.x;
    const int m0 = blockIdx.x * 16;
    const int n0 = blockIdx.y * 64;
    const v8f vz = {0.f,0.f,0.f,0.f,0.f,0.f,0.f,0.f};
    v8f c1[1][4], c2[1][4];
#pragma unroll
    for (int nt = 0; nt < 4; ++nt) { c1[0][nt] = vz; c2[0][nt] = vz; }
    gemm_run<1, 4, true>(xn, xn, 128, pin, gin, 128, m0, n0, 128, lane, c1, c2);
#pragma unroll
    for (int nt = 0; nt < 4; ++nt) {
        int nn = n0 + nt * 16 + (lane & 15);
        u16* dst = (nn < 128) ? aT : bT;
        int d = nn & 127;
#pragma unroll
        for (int r = 0; r < 8; ++r) {
            int mm = m0 + ((lane >> 4) * 8) + r;
            float v = c1[0][nt][r] * sigm(c2[0][nt][r]) * mask[mm];
            int idx = SWZ ? (((mm & 255) << 8) | (mm >> 8)) : mm;   // incoming: [d][k][i]
            dst[(size_t)d * 65536 + idx] = f2bf(v);
        }
    }
}

// per-channel triangle einsum: o[i,j,d] = sum_k a_t[d][i][k] * b_t[d][j][k]
__global__ __launch_bounds__(32) void k_tri_gemm(const u16* __restrict__ aT,
                                                 const u16* __restrict__ bT,
                                                 float* __restrict__ o) {
    const int lane = threadIdx.x;
    const int m0 = blockIdx.x * 32;
    const int n0 = blockIdx.y * 64;
    const int d  = blockIdx.z;
    const u16* A = aT + (size_t)d * 65536;
    const u16* W = bT + (size_t)d * 65536;
    const v8f vz = {0.f,0.f,0.f,0.f,0.f,0.f,0.f,0.f};
    v8f c1[2][4];
#pragma unroll
    for (int mt = 0; mt < 2; ++mt)
#pragma unroll
        for (int nt = 0; nt < 4; ++nt) c1[mt][nt] = vz;
    gemm_run<2, 4, false>(A, A, 256, W, W, 256, m0, n0, 256, lane, c1, c1);
#pragma unroll
    for (int mt = 0; mt < 2; ++mt)
#pragma unroll
        for (int nt = 0; nt < 4; ++nt) {
            int nn = n0 + nt * 16 + (lane & 15);
#pragma unroll
            for (int r = 0; r < 8; ++r) {
                int mm = m0 + mt * 16 + ((lane >> 4) * 8) + r;
                o[((size_t)mm * 256 + nn) * 128 + d] = c1[mt][nt][r];
            }
        }
}

// tri-mul output: z = resid + (o_ln@p_out^T) * sigmoid(xn@g_out^T)
__global__ __launch_bounds__(32) void k_gemm_gate_out(const u16* __restrict__ oln,
                                                      const u16* __restrict__ xn,
                                                      const u16* __restrict__ pout,
                                                      const u16* __restrict__ gout,
                                                      const float* __restrict__ resid,
                                                      float* __restrict__ zout) {
    const int lane = threadIdx.x;
    const int m0 = blockIdx.x * 16;
    const int n0 = blockIdx.y * 64;
    const v8f vz = {0.f,0.f,0.f,0.f,0.f,0.f,0.f,0.f};
    v8f c1[1][4], c2[1][4];
#pragma unroll
    for (int nt = 0; nt < 4; ++nt) { c1[0][nt] = vz; c2[0][nt] = vz; }
    gemm_run<1, 4, true>(oln, xn, 128, pout, gout, 128, m0, n0, 128, lane, c1, c2);
#pragma unroll
    for (int nt = 0; nt < 4; ++nt) {
        int nn = n0 + nt * 16 + (lane & 15);
#pragma unroll
        for (int r = 0; r < 8; ++r) {
            int mm = m0 + ((lane >> 4) * 8) + r;
            size_t a = (size_t)mm * 128 + nn;
            zout[a] = resid[a] + c1[0][nt][r] * sigm(c2[0][nt][r]);
        }
    }
}

// 4 projections (q,k,v,g) selected by blockIdx.z, bf16 out
__global__ __launch_bounds__(32) void k_gemm_bf16x4(const u16* __restrict__ xn,
                                                    const u16* w0, const u16* w1,
                                                    const u16* w2, const u16* w3,
                                                    u16* o0, u16* o1, u16* o2, u16* o3) {
    const int lane = threadIdx.x;
    const int m0 = blockIdx.x * 32;
    const int n0 = blockIdx.y * 64;
    const u16* W; u16* O;
    switch (blockIdx.z) {
        case 0:  W = w0; O = o0; break;
        case 1:  W = w1; O = o1; break;
        case 2:  W = w2; O = o2; break;
        default: W = w3; O = o3; break;
    }
    const v8f vz = {0.f,0.f,0.f,0.f,0.f,0.f,0.f,0.f};
    v8f c1[2][4];
#pragma unroll
    for (int mt = 0; mt < 2; ++mt)
#pragma unroll
        for (int nt = 0; nt < 4; ++nt) c1[mt][nt] = vz;
    gemm_run<2, 4, false>(xn, xn, 128, W, W, 128, m0, n0, 128, lane, c1, c1);
#pragma unroll
    for (int mt = 0; mt < 2; ++mt)
#pragma unroll
        for (int nt = 0; nt < 4; ++nt) {
            int nn = n0 + nt * 16 + (lane & 15);
#pragma unroll
            for (int r = 0; r < 8; ++r) {
                int mm = m0 + mt * 16 + ((lane >> 4) * 8) + r;
                O[(size_t)mm * 128 + nn] = f2bf(c1[mt][nt][r]);
            }
        }
}

// transition hidden half: h = silu(xn@fc1s^T) * (xn@fc2s^T), bf16 [M,256]
__global__ __launch_bounds__(32) void k_gemm_swiglu(const u16* __restrict__ xn,
                                                    const u16* __restrict__ w1,
                                                    const u16* __restrict__ w2,
                                                    u16* __restrict__ hbuf) {
    const int lane = threadIdx.x;
    const int m0 = blockIdx.x * 16;
    const int n0 = blockIdx.y * 64;
    const v8f vz = {0.f,0.f,0.f,0.f,0.f,0.f,0.f,0.f};
    v8f c1[1][4], c2[1][4];
#pragma unroll
    for (int nt = 0; nt < 4; ++nt) { c1[0][nt] = vz; c2[0][nt] = vz; }
    gemm_run<1, 4, true>(xn, xn, 128, w1, w2, 128, m0, n0, 128, lane, c1, c2);
#pragma unroll
    for (int nt = 0; nt < 4; ++nt) {
        int nn = n0 + nt * 16 + (lane & 15);
#pragma unroll
        for (int r = 0; r < 8; ++r) {
            int mm = m0 + ((lane >> 4) * 8) + r;
            float x = c1[0][nt][r];
            hbuf[(size_t)mm * 256 + nn] = f2bf(x * sigm(x) * c2[0][nt][r]);
        }
    }
}

// z += A@W^T  (TRANS: z[j,i] += delta[i,j], for the ending-node attention)
template<bool TRANS>
__global__ __launch_bounds__(32) void k_gemm_resid(const u16* __restrict__ Aa,
                                                   const u16* __restrict__ Wt,
                                                   int K, int lda, int ldw,
                                                   float* __restrict__ zio) {
    const int lane = threadIdx.x;
    const int m0 = blockIdx.x * 32;
    const int n0 = blockIdx.y * 64;
    const v8f vz = {0.f,0.f,0.f,0.f,0.f,0.f,0.f,0.f};
    v8f c1[2][4];
#pragma unroll
    for (int mt = 0; mt < 2; ++mt)
#pragma unroll
        for (int nt = 0; nt < 4; ++nt) c1[mt][nt] = vz;
    gemm_run<2, 4, false>(Aa, Aa, lda, Wt, Wt, ldw, m0, n0, K, lane, c1, c1);
#pragma unroll
    for (int mt = 0; mt < 2; ++mt)
#pragma unroll
        for (int nt = 0; nt < 4; ++nt) {
            int nn = n0 + nt * 16 + (lane & 15);
#pragma unroll
            for (int r = 0; r < 8; ++r) {
                int mm = m0 + mt * 16 + ((lane >> 4) * 8) + r;
                size_t a = TRANS ? ((size_t)(((mm & 255) << 8) | (mm >> 8)) * 128 + nn)
                                 : ((size_t)mm * 128 + nn);
                zio[a] = zio[a] + c1[mt][nt][r];
            }
        }
}

// ------------------------------------------------------------------ fused triangle attention
// one wave per (q-tile16, head, row i): S=Q*K^T (WMMA), +bias +mask, softmax, O=P*V (WMMA), gate
template<bool MTRANS>
__global__ __launch_bounds__(32) void k_attn(const u16* __restrict__ qb, const u16* __restrict__ kb,
                                             const u16* __restrict__ vb, const u16* __restrict__ gb,
                                             const float* __restrict__ tb, const float* __restrict__ mask,
                                             u16* __restrict__ ao) {
    const int L  = threadIdx.x;
    const int q0 = blockIdx.x * 16;
    const int h  = blockIdx.y;
    const int i  = blockIdx.z;
    const int N = 256, HW = 128;
    const int hoff = h * 32;

    __shared__ __align__(16) u16 Pl[16 * 256];

    const int koff = (L >> 4) * 16;
    const int kl = (L & 15);

    // Q A-frag
    const u16* qp = qb + (size_t)(i * N + q0 + kl) * HW + hoff + koff;
    FragU aq; aq.u[0] = *(const uint4*)qp; aq.u[1] = *(const uint4*)(qp + 8);

    const v8f vz = {0.f,0.f,0.f,0.f,0.f,0.f,0.f,0.f};
    v8f S[16];
#pragma unroll
    for (int kt = 0; kt < 16; ++kt) {
        const u16* kp = kb + (size_t)(i * N + kt * 16 + kl) * HW + hoff + ((L >> 4) * 16);
        FragU bk; bk.u[0] = *(const uint4*)kp; bk.u[1] = *(const uint4*)(kp + 8);
        S[kt] = wmma_bf(aq.v, bk.v, vz);
    }

    const int mrow = q0 + ((L >> 4) * 8);
    const float scale = 0.1767766952966369f;   // 1/sqrt(32)
    float rmax[8], rsum[8];
#pragma unroll
    for (int r = 0; r < 8; ++r) rmax[r] = -3.0e38f;
#pragma unroll
    for (int kt = 0; kt < 16; ++kt) {
#pragma unroll
        for (int r = 0; r < 8; ++r) {
            int kk = kt * 16 + kl;
            float mv = MTRANS ? mask[kk * N + i] : mask[i * N + kk];
            float s = S[kt][r] * scale + tb[(h * N + mrow + r) * N + kk] + (mv - 1.0f) * 1.0e9f;
            S[kt][r] = s;
            rmax[r] = fmaxf(rmax[r], s);
        }
    }
#pragma unroll
    for (int r = 0; r < 8; ++r) {
        rmax[r] = fmaxf(rmax[r], __shfl_xor(rmax[r], 1, 32));
        rmax[r] = fmaxf(rmax[r], __shfl_xor(rmax[r], 2, 32));
        rmax[r] = fmaxf(rmax[r], __shfl_xor(rmax[r], 4, 32));
        rmax[r] = fmaxf(rmax[r], __shfl_xor(rmax[r], 8, 32));
        rsum[r] = 0.f;
    }
#pragma unroll
    for (int kt = 0; kt < 16; ++kt) {
#pragma unroll
        for (int r = 0; r < 8; ++r) {
            float p = __expf(S[kt][r] - rmax[r]);
            rsum[r] += p;
            Pl[(((L >> 4) * 8) + r) * 256 + kt * 16 + kl] = f2bf(p);
        }
    }
#pragma unroll
    for (int r = 0; r < 8; ++r) {
        rsum[r] += __shfl_xor(rsum[r], 1, 32);
        rsum[r] += __shfl_xor(rsum[r], 2, 32);
        rsum[r] += __shfl_xor(rsum[r], 4, 32);
        rsum[r] += __shfl_xor(rsum[r], 8, 32);
    }
    __syncthreads();

    v8f O0 = vz, O1 = vz;
#pragma unroll
    for (int kc = 0; kc < 8; ++kc) {
        const u16* pl = &Pl[kl * 256 + kc * 32 + koff];
        FragU ap; ap.u[0] = *(const uint4*)pl; ap.u[1] = *(const uint4*)(pl + 8);
        int krow0 = kc * 32 + ((L >> 4) * 16);
#pragma unroll
        for (int nt = 0; nt < 2; ++nt) {
            union { u16 s[16]; v16bf v; } bv;
            int wcol = hoff + nt * 16 + kl;
#pragma unroll
            for (int t = 0; t < 16; ++t)
                bv.s[t] = vb[(size_t)(i * N + krow0 + t) * HW + wcol];
            if (nt == 0) O0 = wmma_bf(ap.v, bv.v, O0);
            else         O1 = wmma_bf(ap.v, bv.v, O1);
        }
    }

#pragma unroll
    for (int r = 0; r < 8; ++r) {
        float inv = 1.0f / rsum[r];
        int mq = mrow + r;
#pragma unroll
        for (int nt = 0; nt < 2; ++nt) {
            int wcol = hoff + nt * 16 + kl;
            float gv = sigm(bf2f(gb[(size_t)(i * N + mq) * HW + wcol]));
            float ov = (nt == 0 ? O0[r] : O1[r]) * inv * gv;
            ao[(size_t)(i * N + mq) * HW + wcol] = f2bf(ov);
        }
    }
}

// ------------------------------------------------------------------ host orchestration
extern "C" void kernel_launch(void* const* d_in, const int* in_sizes, int n_in,
                              void* d_out, int out_size, void* d_ws, size_t ws_size,
                              hipStream_t stream) {
    (void)in_sizes; (void)n_in; (void)out_size; (void)ws_size;
    const float* z_in  = (const float*)d_in[0];
    const float* maskp = (const float*)d_in[1];
    float* z = (float*)d_out;
    char* ws = (char*)d_ws;
    const size_t M = 65536;   // N*N

    size_t off = 0;
    auto alloc = [&](size_t bytes) { size_t r = off; off += (bytes + 255) & ~(size_t)255; return r; };

    // weight bf16 slots: {d_in index, elem count}
    struct Slot { int idx; int n; size_t off; };
    Slot sl[23] = {
        {4, 32768, 0}, {5, 32768, 0}, {8, 16384, 0}, {9, 16384, 0},       // tmo
        {12, 32768, 0}, {13, 32768, 0}, {16, 16384, 0}, {17, 16384, 0},   // tmi
        {20, 512, 0}, {21, 16384, 0}, {22, 16384, 0}, {23, 16384, 0}, {24, 16384, 0}, {25, 16384, 0}, // tas
        {28, 512, 0}, {29, 16384, 0}, {30, 16384, 0}, {31, 16384, 0}, {32, 16384, 0}, {33, 16384, 0}, // tae
        {36, 65536, 0}, {37, 65536, 0}, {38, 65536, 0},                   // tz fc1,fc2,fc3
    };
    for (int s = 0; s < 23; ++s) sl[s].off = alloc((size_t)sl[s].n * 2);

    size_t xnO = alloc(M * 128 * 2);
    size_t aO  = alloc(M * 128 * 2);   // a_t / q
    size_t bO  = alloc(M * 128 * 2);   // b_t / k
    size_t cO  = alloc(M * 128 * 2);   // o_ln / v
    size_t dO  = alloc(M * 128 * 2);   // g
    size_t eO  = alloc(M * 128 * 2);   // ao
    size_t oO  = alloc(M * 128 * 4);   // o fp32 / zt fp32 / h bf16[M,256]
    size_t tbO = alloc(4 * M * 4);

    auto U = [&](size_t o) -> u16*  { return (u16*)(ws + o); };
    auto Fp = [&](size_t o) -> float* { return (float*)(ws + o); };
    auto IN = [&](int i) -> const float* { return (const float*)d_in[i]; };

    for (int s = 0; s < 23; ++s)
        k_f2bf<<<(sl[s].n + 255) / 256, 256, 0, stream>>>(IN(sl[s].idx), U(sl[s].off), sl[s].n);

    u16 *tmo_pin = U(sl[0].off), *tmo_gin = U(sl[1].off), *tmo_pout = U(sl[2].off), *tmo_gout = U(sl[3].off);
    u16 *tmi_pin = U(sl[4].off), *tmi_gin = U(sl[5].off), *tmi_pout = U(sl[6].off), *tmi_gout = U(sl[7].off);
    u16 *tas_wb = U(sl[8].off), *tas_wq = U(sl[9].off), *tas_wk = U(sl[10].off),
        *tas_wv = U(sl[11].off), *tas_wg = U(sl[12].off), *tas_wo = U(sl[13].off);
    u16 *tae_wb = U(sl[14].off), *tae_wq = U(sl[15].off), *tae_wk = U(sl[16].off),
        *tae_wv = U(sl[17].off), *tae_wg = U(sl[18].off), *tae_wo = U(sl[19].off);
    u16 *fc1 = U(sl[20].off), *fc2 = U(sl[21].off), *fc3 = U(sl[22].off);

    u16 *xn = U(xnO), *aT = U(aO), *bT = U(bO), *cB = U(cO), *dB = U(dO), *eB = U(eO);
    float *obuf = Fp(oO), *tb = Fp(tbO);

    // ---- sublayer 1: triangle multiplicative, outgoing
    k_ln<<<M, 128, 0, stream>>>(z_in, IN(2), IN(3), xn);
    k_gemm_trimul_in<0><<<dim3(4096, 4), 32, 0, stream>>>(xn, tmo_pin, tmo_gin, maskp, aT, bT);
    k_tri_gemm<<<dim3(8, 4, 128), 32, 0, stream>>>(aT, bT, obuf);
    k_ln<<<M, 128, 0, stream>>>(obuf, IN(6), IN(7), cB);
    k_gemm_gate_out<<<dim3(4096, 2), 32, 0, stream>>>(cB, xn, tmo_pout, tmo_gout, z_in, z);

    // ---- sublayer 2: triangle multiplicative, incoming
    k_ln<<<M, 128, 0, stream>>>(z, IN(10), IN(11), xn);
    k_gemm_trimul_in<1><<<dim3(4096, 4), 32, 0, stream>>>(xn, tmi_pin, tmi_gin, maskp, aT, bT);
    k_tri_gemm<<<dim3(8, 4, 128), 32, 0, stream>>>(aT, bT, obuf);
    k_ln<<<M, 128, 0, stream>>>(obuf, IN(14), IN(15), cB);
    k_gemm_gate_out<<<dim3(4096, 2), 32, 0, stream>>>(cB, xn, tmi_pout, tmi_gout, z, z);

    // ---- sublayer 3: triangle attention, starting node
    k_ln<<<M, 128, 0, stream>>>(z, IN(18), IN(19), xn);
    k_gemm_bf16x4<<<dim3(2048, 2, 4), 32, 0, stream>>>(xn, tas_wq, tas_wk, tas_wv, tas_wg, aT, bT, cB, dB);
    k_tb<<<256, 256, 0, stream>>>(xn, tas_wb, tb);
    k_attn<false><<<dim3(16, 4, 256), 32, 0, stream>>>(aT, bT, cB, dB, tb, maskp, eB);
    k_gemm_resid<false><<<dim3(2048, 2), 32, 0, stream>>>(eB, tas_wo, 128, 128, 128, z);

    // ---- sublayer 4: triangle attention, ending node (on z^T)
    k_transpose<<<M, 128, 0, stream>>>(z, obuf);
    k_ln<<<M, 128, 0, stream>>>(obuf, IN(26), IN(27), xn);
    k_gemm_bf16x4<<<dim3(2048, 2, 4), 32, 0, stream>>>(xn, tae_wq, tae_wk, tae_wv, tae_wg, aT, bT, cB, dB);
    k_tb<<<256, 256, 0, stream>>>(xn, tae_wb, tb);
    k_attn<true><<<dim3(16, 4, 256), 32, 0, stream>>>(aT, bT, cB, dB, tb, maskp, eB);
    k_gemm_resid<true><<<dim3(2048, 2), 32, 0, stream>>>(eB, tae_wo, 128, 128, 128, z);

    // ---- sublayer 5: SwiGLU transition (two hidden halves)
    k_ln<<<M, 128, 0, stream>>>(z, IN(34), IN(35), xn);
    u16* hbuf = (u16*)obuf;
    for (int pass = 0; pass < 2; ++pass) {
        k_gemm_swiglu<<<dim3(4096, 4), 32, 0, stream>>>(xn, fc1 + pass * 256 * 128, fc2 + pass * 256 * 128, hbuf);
        k_gemm_resid<false><<<dim3(2048, 2), 32, 0, stream>>>(hbuf, fc3 + pass * 256, 256, 256, 512, z);
    }
}
